// CTC_model_63419487093322
// MI455X (gfx1250) — compile-verified, matrix operand order checked by reading, not run on hardware
//
#include <hip/hip_runtime.h>

// ---------------------------------------------------------------------------
// CTC model for MI455X (gfx1250, wave32, WMMA):
//   logits = softmax(feat[81920,512] @ W[512,78] + b)   (bf16x3 split WMMA)
//   best   = argmax(logits)  -> CTC greedy decode -> [1024,30]
// Memory-bound problem (~194MB @ 23.3TB/s => ~8.3us floor); WMMA compute is
// ~20 GF-equivalent even with the 3x fp32-emulation split => compute is free.
// ---------------------------------------------------------------------------

typedef __attribute__((ext_vector_type(16))) __bf16 v16bf;
typedef __attribute__((ext_vector_type(8)))  float  v8f;

#define VOCAB_C   78        // VOCAB+1 classes
#define BLANK_ID  77
#define D_DIM     512
#define T_DIM     80
#define B_DIM     1024
#define M_ROWS    (B_DIM * T_DIM)          // 81920
#define PRED_LEN  30
#define NT        5                        // 5 N-tiles of 16 (80 cols, 78 valid)
#define KC        16                       // 16 K-chunks of 32
#define FRAG_ELEMS (NT * KC * 32 * 16)     // 40960 bf16 per (hi|lo) array

__device__ __forceinline__ __bf16 bf16_trunc(float f) {
  unsigned u = __builtin_bit_cast(unsigned, f);
  return __builtin_bit_cast(__bf16, (unsigned short)(u >> 16));
}

// f = hi + lo exactly for hi; lo truncated to bf16 (error ~2^-16 relative).
__device__ __forceinline__ void splitf(float f, __bf16& hi, __bf16& lo) {
  unsigned u = __builtin_bit_cast(unsigned, f);
  float fh = __builtin_bit_cast(float, u & 0xFFFF0000u);
  hi = __builtin_bit_cast(__bf16, (unsigned short)(u >> 16));
  lo = bf16_trunc(f - fh);
}

// ---------------------------------------------------------------------------
// Kernel 0: pack W[512,78] into wave32 WMMA B-fragment layout, hi/lo bf16.
// B is KxN per chunk (32x16, 16-bit): lane p -> column N = tile*16 + (p&15);
// half = p>>4; bf16 elem j -> K = kc*32 + (j<8 ? half*8+j : 16+half*8+(j-8)).
// Columns >= 78 are zero-padded.
// ---------------------------------------------------------------------------
__global__ void pack_w_kernel(const float* __restrict__ W,
                              __bf16* __restrict__ fragHi,
                              __bf16* __restrict__ fragLo) {
  int idx = blockIdx.x * blockDim.x + threadIdx.x;
  if (idx >= FRAG_ELEMS) return;
  int j    = idx & 15;
  int p    = (idx >> 4) & 31;
  int kc   = (idx >> 9) & 15;
  int t    = idx >> 13;
  int half = p >> 4;
  int n    = t * 16 + (p & 15);
  int jj   = (j < 8) ? (half * 8 + j) : (16 + half * 8 + (j - 8));
  int k    = kc * 32 + jj;
  float f  = (n < VOCAB_C) ? W[k * VOCAB_C + n] : 0.0f;
  __bf16 hi, lo;
  splitf(f, hi, lo);
  fragHi[idx] = hi;
  fragLo[idx] = lo;
}

// ---------------------------------------------------------------------------
// Kernel 1: fused GEMM (bf16x3 WMMA) + bias + softmax + argmax.
// One wave -> one 16-row x 80-col tile. 8 waves / block, 640 blocks.
// A layout (16-bit 16x32): lane half 0 -> K {0..7,16..23}, half 1 -> {8..15,24..31},
// row M = m0 + (lane&15). C/D: reg r, lane -> (M = r + 8*half, N = lane&15).
// ---------------------------------------------------------------------------
__global__ void gemm_softmax_argmax_kernel(const float* __restrict__ feat,
                                           const float* __restrict__ bias,
                                           const __bf16* __restrict__ fragHi,
                                           const __bf16* __restrict__ fragLo,
                                           float* __restrict__ probs,
                                           int* __restrict__ best) {
  const int lane = threadIdx.x & 31;
  const int wave = threadIdx.x >> 5;
  const int half = lane >> 4;
  const int lp   = lane & 15;
  const int m0   = blockIdx.x * 128 + wave * 16;

  v8f acc[NT] = {};

  const float* rowp = feat + (size_t)(m0 + lp) * D_DIM;
  const v16bf* bHi  = (const v16bf*)fragHi;
  const v16bf* bLo  = (const v16bf*)fragLo;

#pragma unroll 1
  for (int kc = 0; kc < KC; ++kc) {
    const int k0 = kc * 32;
    // Each feat element is loaded by exactly one lane exactly once.
    const float4* pa = (const float4*)(rowp + k0 + half * 8);
    const float4* pb = (const float4*)(rowp + k0 + 16 + half * 8);
    float4 g0 = pa[0], g1 = pa[1], g2 = pb[0], g3 = pb[1];
    if (kc + 1 < KC)  // pull next A chunk toward the WGP while we compute
      __builtin_prefetch(rowp + (kc + 1) * 32 + half * 8, 0, 3);

    float av[16] = {g0.x, g0.y, g0.z, g0.w, g1.x, g1.y, g1.z, g1.w,
                    g2.x, g2.y, g2.z, g2.w, g3.x, g3.y, g3.z, g3.w};
    v16bf ahi, alo;
#pragma unroll
    for (int j = 0; j < 16; ++j) {
      __bf16 h, l;
      splitf(av[j], h, l);
      ahi[j] = h;
      alo[j] = l;
    }

#pragma unroll
    for (int t = 0; t < NT; ++t) {
      const int fidx = (t * KC + kc) * 32 + lane;
      const v16bf bh = bHi[fidx];
      const v16bf bl = bLo[fidx];
      // fp32-accurate product: ah*bh + ah*bl + al*bh (al*bl negligible)
      acc[t] = __builtin_amdgcn_wmma_f32_16x16x32_bf16(
          false, ahi, false, bh, (short)0, acc[t], false, false);
      acc[t] = __builtin_amdgcn_wmma_f32_16x16x32_bf16(
          false, ahi, false, bl, (short)0, acc[t], false, false);
      acc[t] = __builtin_amdgcn_wmma_f32_16x16x32_bf16(
          false, alo, false, bh, (short)0, acc[t], false, false);
    }
  }

  // ---- bias ----
#pragma unroll
  for (int t = 0; t < NT; ++t) {
    const int n = t * 16 + lp;
    const float bv = (n < VOCAB_C) ? bias[n] : 0.0f;
#pragma unroll
    for (int r = 0; r < 8; ++r) acc[t][r] += bv;
  }

  // ---- per-row softmax + argmax (row lives in one 16-lane half) ----
#pragma unroll
  for (int r = 0; r < 8; ++r) {
    float vmax = -3.4e38f;
    int   vidx = 0x7fffffff;
#pragma unroll
    for (int t = 0; t < NT; ++t) {
      const int n = t * 16 + lp;
      const float v = acc[t][r];
      if (n < VOCAB_C && (v > vmax || (v == vmax && n < vidx))) {
        vmax = v;
        vidx = n;
      }
    }
#pragma unroll
    for (int off = 1; off < 16; off <<= 1) {  // stays within the 16-lane half
      const float ov = __shfl_xor(vmax, off, 32);
      const int   oi = __shfl_xor(vidx, off, 32);
      if (ov > vmax || (ov == vmax && oi < vidx)) {
        vmax = ov;
        vidx = oi;
      }
    }
    float s = 0.0f;
    float e[NT];
#pragma unroll
    for (int t = 0; t < NT; ++t) {
      const int n = t * 16 + lp;
      const float v = (n < VOCAB_C) ? __expf(acc[t][r] - vmax) : 0.0f;
      e[t] = v;
      s += v;
    }
#pragma unroll
    for (int off = 1; off < 16; off <<= 1) s += __shfl_xor(s, off, 32);
    const float inv = 1.0f / s;

    const int row = m0 + half * 8 + r;
    float* outp = probs + (size_t)row * VOCAB_C;
#pragma unroll
    for (int t = 0; t < NT; ++t) {
      const int n = t * 16 + lp;
      if (n < VOCAB_C) outp[n] = e[t] * inv;
    }
    if (lp == 0) best[row] = vidx;
  }
}

// ---------------------------------------------------------------------------
// Kernel 2: CTC greedy decode, one thread per batch row.
// keep = (best[t] != BLANK) && (best[t] != best[t-1]); overflow dropped;
// pad to PRED_LEN with -1; emitted as float (d_out is float).
// ---------------------------------------------------------------------------
__global__ void ctc_decode_kernel(const int* __restrict__ best,
                                  float* __restrict__ lab) {
  const int b = blockIdx.x * blockDim.x + threadIdx.x;
  if (b >= B_DIM) return;
  const int* row = best + b * T_DIM;
  float out[PRED_LEN];
#pragma unroll
  for (int i = 0; i < PRED_LEN; ++i) out[i] = -1.0f;
  int prev = -1, cnt = 0;
  for (int t = 0; t < T_DIM; ++t) {
    const int v = row[t];
    if (v != BLANK_ID && v != prev && cnt < PRED_LEN) out[cnt++] = (float)v;
    prev = v;
  }
  float* op = lab + b * PRED_LEN;
#pragma unroll
  for (int i = 0; i < PRED_LEN; ++i) op[i] = out[i];
}

// ---------------------------------------------------------------------------
extern "C" void kernel_launch(void* const* d_in, const int* in_sizes, int n_in,
                              void* d_out, int out_size, void* d_ws,
                              size_t ws_size, hipStream_t stream) {
  const float* feat = (const float*)d_in[0];  // [1024,80,512] f32
  const float* W    = (const float*)d_in[1];  // [512,78] f32
  const float* bias = (const float*)d_in[2];  // [78] f32
  // d_in[3] = y (unused), d_in[4] = times (=80, compile-time constant here)

  __bf16* fragHi = (__bf16*)d_ws;
  __bf16* fragLo = fragHi + FRAG_ELEMS;
  int*    best   = (int*)((char*)d_ws + 2 * (size_t)FRAG_ELEMS * sizeof(__bf16));

  float* probs = (float*)d_out;                           // [81920,78]
  float* lab   = probs + (size_t)M_ROWS * VOCAB_C;        // [1024,30] as float

  pack_w_kernel<<<(FRAG_ELEMS + 255) / 256, 256, 0, stream>>>(W, fragHi, fragLo);
  gemm_softmax_argmax_kernel<<<M_ROWS / 128, 256, 0, stream>>>(
      feat, bias, fragHi, fragLo, probs, best);
  ctc_decode_kernel<<<(B_DIM + 255) / 256, 256, 0, stream>>>(best, lab);
}
